// GCN_38104949850570
// MI455X (gfx1250) — matrix operand order repaired and checked by reading, not compiled, not run on hardware
//
#include <hip/hip_runtime.h>
#include <hip/hip_bf16.h>

typedef _Float16 h16;
typedef __attribute__((ext_vector_type(16))) _Float16 v16h;
typedef __attribute__((ext_vector_type(8)))  _Float16 v8h;
typedef __attribute__((ext_vector_type(8)))  float    v8f;

__device__ __forceinline__ void atomAddF(float* p, float v) {
    __hip_atomic_fetch_add(p, v, __ATOMIC_RELAXED, __HIP_MEMORY_SCOPE_AGENT);
}

// ---------------------------------------------------------------- utilities
__global__ void k_zero(float* __restrict__ p, int n) {
    int i = blockIdx.x * blockDim.x + threadIdx.x;
    if (i < n) p[i] = 0.0f;
}

__global__ void k_f32_to_f16(const float* __restrict__ x, h16* __restrict__ y, int n) {
    int i = blockIdx.x * blockDim.x + threadIdx.x;
    if (i < n) y[i] = (h16)x[i];
}

// W is [K][64] row-major (in,out); produce Wt[64][K] f16
__global__ void k_transpose_w(const float* __restrict__ W, h16* __restrict__ Wt, int K) {
    int i = blockIdx.x * blockDim.x + threadIdx.x;
    if (i < K * 64) {
        int n = i / K, k = i % K;
        Wt[i] = (h16)W[k * 64 + n];
    }
}

__global__ void k_deg(const int* __restrict__ col, float* __restrict__ deg, int E) {
    int e = blockIdx.x * blockDim.x + threadIdx.x;
    if (e < E) atomAddF(&deg[col[e]], 1.0f);
}

__global__ void k_dis(float* __restrict__ deg_dis, int n) {
    int i = blockIdx.x * blockDim.x + threadIdx.x;
    if (i < n) deg_dis[i] = rsqrtf(deg_dis[i] + 1.0f);  // +1 for self loop
}

// ---------------------------------------------------------------- WMMA GEMM
// One block = 128 threads = 4 waves, covering one 16-row tile of X and
// all 4 column tiles of H=64. Each wave computes D16x16 for BOTH W and RW
// (shared A fragment). X is [nrows][K] f16 row-major; Wt/RWt are [64][K] f16.
template <int K>
__global__ void k_gemm_dual(const h16* __restrict__ X,
                            const h16* __restrict__ Wt,
                            const h16* __restrict__ RWt,
                            float* __restrict__ Hh,
                            float* __restrict__ Rr,
                            int nrows) {
    const int lane = threadIdx.x & 31;
    const int wave = threadIdx.x >> 5;              // n-tile 0..3
    const int row0 = blockIdx.x * 16;
    const int m    = lane & 15;
    const int kbA  = (lane >> 4) * 16;              // A: one run of 16 halves
    const int kbB  = (lane >> 4) * 16;              // B: one run of 16 halves
    const int n    = (lane & 15) + wave * 16;

    const h16* xrow = X   + (size_t)(row0 + m) * K;
    const h16* w0   = Wt  + (size_t)n * K;
    const h16* w1   = RWt + (size_t)n * K;

    v8f c0 = {};
    v8f c1 = {};
#pragma unroll
    for (int kk = 0; kk < K; kk += 32) {
        v8h alo = *(const v8h*)(xrow + kk + kbA);
        v8h ahi = *(const v8h*)(xrow + kk + kbA + 8);
        v16h a;
#pragma unroll
        for (int i = 0; i < 8; ++i) { a[i] = alo[i]; a[i + 8] = ahi[i]; }
        v16h b0 = *(const v16h*)(w0 + kk + kbB);
        v16h b1 = *(const v16h*)(w1 + kk + kbB);
        c0 = __builtin_amdgcn_wmma_f32_16x16x32_f16(false, a, false, b0, (short)0, c0, false, false);
        c1 = __builtin_amdgcn_wmma_f32_16x16x32_f16(false, a, false, b1, (short)0, c1, false, false);
    }

    const int rbase = row0 + (lane >> 4) * 8;
#pragma unroll
    for (int v = 0; v < 8; ++v) {
        int r = rbase + v;
        if (r < nrows) {
            Hh[(size_t)r * 64 + n] = c0[v];
            Rr[(size_t)r * 64 + n] = c1[v];
        }
    }
}

// acc = r + rb + b + h * dis^2   (self-loop term), in place over r
__global__ void k_init_acc(const float* __restrict__ h, float* __restrict__ racc,
                           const float* __restrict__ b, const float* __restrict__ rb,
                           const float* __restrict__ dis, int total) {
    int idx = blockIdx.x * blockDim.x + threadIdx.x;
    if (idx < total) {
        int i = idx >> 6, f = idx & 63;
        float s = dis[i];
        racc[idx] = racc[idx] + b[f] + rb[f] + h[idx] * s * s;
    }
}

// acc[col] += h[row] * dis[row] * dis[col]; 64 lanes per edge (wave-uniform edge)
__global__ void k_edge_scatter(const int* __restrict__ row, const int* __restrict__ col,
                               const float* __restrict__ h, const float* __restrict__ dis,
                               float* __restrict__ acc, int total) {
    int idx = blockIdx.x * blockDim.x + threadIdx.x;
    if (idx < total) {
        int e = idx >> 6, f = idx & 63;
        int rs = row[e], ct = col[e];
        float v = h[(size_t)rs * 64 + f] * dis[rs] * dis[ct];
        atomAddF(&acc[(size_t)ct * 64 + f], v);
    }
}

// one wave32 per node; 2 features/lane; butterfly reduce; optional ReLU
__global__ void k_layernorm(float* __restrict__ acc, h16* __restrict__ x16,
                            const float* __restrict__ g, const float* __restrict__ be,
                            int n, int relu) {
    int lane = threadIdx.x & 31;
    int node = blockIdx.x * (blockDim.x >> 5) + (threadIdx.x >> 5);
    if (node >= n) return;
    float v0 = acc[(size_t)node * 64 + lane];
    float v1 = acc[(size_t)node * 64 + lane + 32];
    float s  = v0 + v1;
    float s2 = v0 * v0 + v1 * v1;
#pragma unroll
    for (int m = 16; m >= 1; m >>= 1) {
        s  += __shfl_xor(s,  m, 32);
        s2 += __shfl_xor(s2, m, 32);
    }
    float mean = s * (1.0f / 64.0f);
    float var  = s2 * (1.0f / 64.0f) - mean * mean;
    float inv  = rsqrtf(var + 1e-5f);
    float y0 = (v0 - mean) * inv * g[lane]      + be[lane];
    float y1 = (v1 - mean) * inv * g[lane + 32] + be[lane + 32];
    if (relu) { y0 = fmaxf(y0, 0.0f); y1 = fmaxf(y1, 0.0f); }
    acc[(size_t)node * 64 + lane]      = y0;
    acc[(size_t)node * 64 + lane + 32] = y1;
    x16[(size_t)node * 64 + lane]      = (h16)y0;
    x16[(size_t)node * 64 + lane + 32] = (h16)y1;
}

__global__ void k_pool(const float* __restrict__ x, const int* __restrict__ batch,
                       float* __restrict__ pooled, float* __restrict__ counts, int total) {
    int idx = blockIdx.x * blockDim.x + threadIdx.x;
    if (idx < total) {
        int i = idx >> 6, f = idx & 63;
        int gi = batch[i];
        atomAddF(&pooled[(size_t)gi * 64 + f], x[idx]);
        if (f == 0) atomAddF(&counts[gi], 1.0f);
    }
}

__global__ void k_final(const float* __restrict__ pooled, const float* __restrict__ counts,
                        const float* __restrict__ lw, const float* __restrict__ lb,
                        float* __restrict__ out, int G) {
    int g = blockIdx.x * blockDim.x + threadIdx.x;
    if (g < G) {
        float c = fmaxf(counts[g], 1.0f);
        float s = 0.0f;
        for (int k = 0; k < 64; ++k) s += pooled[(size_t)g * 64 + k] * lw[k];
        out[g] = s / c + lb[0];
    }
}

// ---------------------------------------------------------------- launcher
extern "C" void kernel_launch(void* const* d_in, const int* in_sizes, int n_in,
                              void* d_out, int out_size, void* d_ws, size_t ws_size,
                              hipStream_t stream) {
    const int N = in_sizes[0] / 128;   // nodes
    const int E = in_sizes[1] / 2;     // edges
    const int G = out_size;            // graphs

    const float* x     = (const float*)d_in[0];
    const int*   ei    = (const int*)d_in[1];
    const int*   row   = ei;           // sources
    const int*   colv  = ei + E;       // targets
    const int*   batch = (const int*)d_in[2];

    const float* Wl[3]  = {(const float*)d_in[3],  (const float*)d_in[5],  (const float*)d_in[7]};
    const float* bl[3]  = {(const float*)d_in[4],  (const float*)d_in[6],  (const float*)d_in[8]};
    const float* RWl[3] = {(const float*)d_in[9],  (const float*)d_in[11], (const float*)d_in[13]};
    const float* rbl[3] = {(const float*)d_in[10], (const float*)d_in[12], (const float*)d_in[14]};
    const float* gl[3]  = {(const float*)d_in[15], (const float*)d_in[17], (const float*)d_in[19]};
    const float* bel[3] = {(const float*)d_in[16], (const float*)d_in[18], (const float*)d_in[20]};
    const float* lw = (const float*)d_in[21];
    const float* lb = (const float*)d_in[22];

    // workspace carve-up (256B aligned slices)
    char* ws = (char*)d_ws;
    size_t off = 0;
    auto take = [&](size_t bytes) -> void* {
        void* p = ws + off;
        off = (off + bytes + 255) & ~(size_t)255;
        return p;
    };
    h16*   x16    = (h16*)  take((size_t)N * 128 * sizeof(h16));
    float* hbuf   = (float*)take((size_t)N * 64 * sizeof(float));
    float* racc   = (float*)take((size_t)N * 64 * sizeof(float));
    float* dis    = (float*)take((size_t)N * sizeof(float));
    float* pooled = (float*)take((size_t)G * 64 * sizeof(float));
    float* counts = (float*)take((size_t)G * sizeof(float));
    h16* Wt[3], *RWt[3];
    const int Ks[3] = {128, 64, 64};
    for (int l = 0; l < 3; ++l) {
        Wt[l]  = (h16*)take((size_t)Ks[l] * 64 * sizeof(h16));
        RWt[l] = (h16*)take((size_t)Ks[l] * 64 * sizeof(h16));
    }
    (void)ws_size; (void)n_in;

    const int T = 256;
    auto blks = [](int n, int t) { return (n + t - 1) / t; };

    // zero accumulators (every call: graph-replay safe)
    k_zero<<<blks(N, T), T, 0, stream>>>(dis, N);
    k_zero<<<blks(G * 64, T), T, 0, stream>>>(pooled, G * 64);
    k_zero<<<blks(G, T), T, 0, stream>>>(counts, G);

    // degrees (with self loops) -> dis = deg^-1/2
    k_deg<<<blks(E, T), T, 0, stream>>>(colv, dis, E);
    k_dis<<<blks(N, T), T, 0, stream>>>(dis, N);

    // f16 copies of x and transposed weights
    k_f32_to_f16<<<blks(N * 128, T), T, 0, stream>>>(x, x16, N * 128);
    for (int l = 0; l < 3; ++l) {
        k_transpose_w<<<blks(Ks[l] * 64, T), T, 0, stream>>>(Wl[l],  Wt[l],  Ks[l]);
        k_transpose_w<<<blks(Ks[l] * 64, T), T, 0, stream>>>(RWl[l], RWt[l], Ks[l]);
    }

    // three GCN layers
    for (int l = 0; l < 3; ++l) {
        int gx = (N + 15) / 16;
        if (Ks[l] == 128)
            k_gemm_dual<128><<<gx, 128, 0, stream>>>(x16, Wt[l], RWt[l], hbuf, racc, N);
        else
            k_gemm_dual<64><<<gx, 128, 0, stream>>>(x16, Wt[l], RWt[l], hbuf, racc, N);

        k_init_acc<<<blks(N * 64, T), T, 0, stream>>>(hbuf, racc, bl[l], rbl[l], dis, N * 64);
        k_edge_scatter<<<blks(E * 64, T), T, 0, stream>>>(row, colv, hbuf, dis, racc, E * 64);
        k_layernorm<<<blks(N, 8), 256, 0, stream>>>(racc, x16, gl[l], bel[l], N, l < 2 ? 1 : 0);
    }

    // global mean pool + readout
    k_pool<<<blks(N * 64, T), T, 0, stream>>>(racc, batch, pooled, counts, N * 64);
    k_final<<<1, G, 0, stream>>>(pooled, counts, lw, lb, (float*)d_out, G);
}
